// DFormerBlock3D_12300786336380
// MI455X (gfx1250) — compile-verified, matrix operand order checked by reading, not run on hardware
//
#include <hip/hip_runtime.h>
#include <hip/hip_bf16.h>
#include <math.h>

// ---------------------------------------------------------------------------
// CDNA5 (gfx1250) bf16 WMMA implementation of the DFormer 3D window block.
// GEMMs use v_wmma_f32_16x16x32_bf16 (wave32); A-tiles staged into LDS with
// GLOBAL_LOAD_ASYNC_TO_LDS_B128 (ASYNCcnt) to bypass the VGPR file.
// ---------------------------------------------------------------------------

typedef __bf16 bf16_t;
typedef __attribute__((ext_vector_type(8)))  __bf16 v8bf;
typedef __attribute__((ext_vector_type(16))) __bf16 v16bf;
typedef __attribute__((ext_vector_type(8)))  float  v8f;

#define TOK   103968     // 2*16*57*57 valid tokens
#define NWIN  1296       // 2 * 8*9*9 windows
#define NPAD  128        // 98 tokens padded to 128
#define CDIM  192
#define HEADS 6
#define DH    32
#define HID   768

// --- fragment load: row-major [rows][stride] source (LDS or global) --------
// A-matrix 16x32 bf16 layout (ISA 7.12.2): lane = half*16 + r holds row r,
// elements 0..7  -> K = k0 + 8*half + e
// elements 8..15 -> K = k0 + 16 + 8*half + (e-8)
// B-matrix (KxN) is loaded from its transpose (N-major rows), same pattern.
__device__ inline v16bf load_frag(const bf16_t* __restrict__ base, int row0,
                                  int stride, int k0) {
  int lane = threadIdx.x & 31;
  int hf   = lane >> 4;
  int r    = lane & 15;
  const bf16_t* p = base + (size_t)(row0 + r) * stride + k0 + 8 * hf;
  v8bf lo = *(const v8bf*)p;
  v8bf hi = *(const v8bf*)(p + 16);
  return __builtin_shufflevector(lo, hi, 0, 1, 2, 3, 4, 5, 6, 7,
                                 8, 9, 10, 11, 12, 13, 14, 15);
}

__device__ inline v8f wmma_bf16(v16bf a, v16bf b, v8f c) {
  return __builtin_amdgcn_wmma_f32_16x16x32_bf16(
      /*neg_a=*/false, a, /*neg_b=*/false, b,
      /*c_mod=*/(short)0, c, /*reuse_a=*/false, /*reuse_b=*/false);
}

// --- async Memory->LDS copy (CDNA5 GLOBAL_LOAD_ASYNC_TO_LDS_B128) ----------
// VDST VGPR carries the per-lane LDS byte address (low 32 bits of the flat
// pointer in the LDS aperture), VADDR the 64-bit global address; tracked by
// ASYNCcnt, drained with s_wait_asynccnt before the workgroup barrier.
__device__ inline void async_copy_b128(bf16_t* lds_dst, const bf16_t* gsrc) {
  unsigned int l = (unsigned int)(uintptr_t)lds_dst;
  asm volatile("global_load_async_to_lds_b128 %0, %1, off"
               :: "v"(l), "v"(gsrc)
               : "memory");
}
__device__ inline void async_wait0() {
  asm volatile("s_wait_asynccnt 0x0" ::: "memory");
}

// ---------------------------------------------------------------------------
// Kernel 1: convert + transpose weights to bf16 (B^T row-major layouts).
// ---------------------------------------------------------------------------
__global__ void k_prep_weights(const float* __restrict__ qkv_w,
                               const float* __restrict__ proj_w,
                               const float* __restrict__ fc1_w,
                               const float* __restrict__ fc2_w,
                               bf16_t* qkvT, bf16_t* projT,
                               bf16_t* fc1T, bf16_t* fc2T) {
  int i = blockIdx.x * 256 + threadIdx.x;
  if (i < 576 * 192) { int n = i / 192, k = i % 192; qkvT[i] = (bf16_t)qkv_w[k * 576 + n]; }
  if (i < 192 * 192) { int n = i / 192, k = i % 192; projT[i] = (bf16_t)proj_w[k * 192 + n]; }
  if (i < 768 * 192) { int n = i / 192, k = i % 192; fc1T[i] = (bf16_t)fc1_w[k * 768 + n]; }
  if (i < 192 * 768) { int n = i / 768, k = i % 768; fc2T[i] = (bf16_t)fc2_w[k * 192 + n]; }
}

// ---------------------------------------------------------------------------
// Kernel 2: LayerNorm1 + window gather -> bf16 xw[NWIN][128][192]
// window order: b-major, then ndi(8), nhi(9), nwi(9); token = gd*49+gh*7+gw.
// ---------------------------------------------------------------------------
__global__ void k_ln1(const float* __restrict__ x, const float* __restrict__ g,
                      const float* __restrict__ b, bf16_t* __restrict__ xw) {
  int win = blockIdx.x;
  int bb  = win / 648;
  int rem = win % 648;
  int ndi = rem / 81; rem %= 81;
  int nhi = rem / 9;
  int nwi = rem % 9;
  int wave = threadIdx.x >> 5, lane = threadIdx.x & 31;

  for (int t = wave; t < NPAD; t += 8) {
    bf16_t* dst = xw + ((size_t)win * NPAD + t) * CDIM;
    bool valid = false;
    size_t src = 0;
    if (t < 98) {
      int gd = t / 49, rr = t % 49, gh = rr / 7, gw = rr % 7;
      int d = ndi * 2 + gd, h = nhi * 7 + gh, w = nwi * 7 + gw;
      if (d < 16 && h < 57 && w < 57) {
        valid = true;
        src = ((((size_t)bb * 16 + d) * 57 + h) * 57 + w) * CDIM;
      }
    }
    if (!valid) {
      for (int c = lane; c < CDIM; c += 32) dst[c] = (bf16_t)0.0f;
    } else {
      float v[6], s = 0.0f;
      for (int j = 0; j < 6; j++) { v[j] = x[src + lane + 32 * j]; s += v[j]; }
      for (int m = 16; m >= 1; m >>= 1) s += __shfl_xor(s, m, 32);
      float mean = s * (1.0f / 192.0f), q = 0.0f;
      for (int j = 0; j < 6; j++) { float d2 = v[j] - mean; q += d2 * d2; }
      for (int m = 16; m >= 1; m >>= 1) q += __shfl_xor(q, m, 32);
      float rstd = rsqrtf(q * (1.0f / 192.0f) + 1e-5f);
      for (int j = 0; j < 6; j++) {
        int c = lane + 32 * j;
        dst[c] = (bf16_t)((v[j] - mean) * rstd * g[c] + b[c]);
      }
    }
  }
}

// ---------------------------------------------------------------------------
// Kernel 3: QKV GEMM per window. grid = NWIN*9 (64-col tiles of the 576 out).
// Outputs scattered into per-head layouts: q[w][h][N][32], k[w][h][N][32],
// vT[w][h][32][N]  (vT transposed so P.V can load B^T row-major).
// ---------------------------------------------------------------------------
__global__ void k_qkv(const bf16_t* __restrict__ xw, const bf16_t* __restrict__ qkvT,
                      const float* __restrict__ qkv_b,
                      bf16_t* __restrict__ qo, bf16_t* __restrict__ ko,
                      bf16_t* __restrict__ vT) {
  __shared__ bf16_t A[NPAD * CDIM];  // 48 KB
  int win = blockIdx.x / 9, nt = blockIdx.x % 9;
  const bf16_t* src = xw + (size_t)win * NPAD * CDIM;
  for (int i = threadIdx.x; i < NPAD * CDIM / 8; i += 256)
    async_copy_b128(A + i * 8, src + i * 8);
  async_wait0();
  __syncthreads();

  int wave = threadIdx.x >> 5, lane = threadIdx.x & 31;
  int hf = lane >> 4, ln = lane & 15;
  int m0 = wave * 16;
  v8f acc[4] = {};
  for (int k0 = 0; k0 < CDIM; k0 += 32) {
    v16bf a = load_frag(A, m0, CDIM, k0);
    for (int j = 0; j < 4; j++) {
      v16bf bf = load_frag(qkvT, nt * 64 + j * 16, CDIM, k0);
      acc[j] = wmma_bf16(a, bf, acc[j]);
    }
  }
  for (int j = 0; j < 4; j++) {
    int n0 = nt * 64 + j * 16;
    for (int e = 0; e < 8; e++) {
      int mrow = m0 + 8 * hf + e;         // token row
      int col  = n0 + ln;                 // 0..575
      float val = acc[j][e] + qkv_b[col];
      int which = col / CDIM;             // 0=q 1=k 2=v (64-col tile | 192)
      int c = col % CDIM;
      int head = c >> 5, dh = c & 31;
      bf16_t bv = (bf16_t)val;
      size_t hb = ((size_t)win * HEADS + head);
      if (which == 0)      qo[(hb * NPAD + mrow) * DH + dh] = bv;
      else if (which == 1) ko[(hb * NPAD + mrow) * DH + dh] = bv;
      else                 vT[(hb * DH + dh) * NPAD + mrow] = bv;
    }
  }
}

// ---------------------------------------------------------------------------
// Kernel 4: fused attention per (window, head). grid = NWIN*HEADS.
// S = q.k^T (single wmma per tile: K = dh = 32), bias + softmax in registers
// (rows live in the 16-lane half -> shfl_xor {8,4,2,1}), P->LDS, O = P.V.
// ---------------------------------------------------------------------------
__global__ void k_attn(const bf16_t* __restrict__ qg, const bf16_t* __restrict__ kg,
                       const bf16_t* __restrict__ vTg, bf16_t* __restrict__ og) {
  __shared__ bf16_t sm[20480];        // 40 KB: P(0..16383) | vT(16384..20479)
  bf16_t* sQ = sm;                    // 128x32 (overlaps P, freed before P writes)
  bf16_t* sK = sm + 4096;             // 128x32
  bf16_t* sP = sm;                    // 128x128
  bf16_t* sV = sm + 16384;            // 32x128

  int win = blockIdx.x / HEADS, head = blockIdx.x % HEADS;
  size_t hb = (size_t)win * HEADS + head;
  const bf16_t* qsrc = qg  + hb * NPAD * DH;
  const bf16_t* ksrc = kg  + hb * NPAD * DH;
  const bf16_t* vsrc = vTg + hb * DH * NPAD;
  for (int i = threadIdx.x; i < 4096 / 8; i += 256) {
    async_copy_b128(sQ + i * 8, qsrc + i * 8);
    async_copy_b128(sK + i * 8, ksrc + i * 8);
    async_copy_b128(sV + i * 8, vsrc + i * 8);
  }
  async_wait0();
  __syncthreads();

  int wave = threadIdx.x >> 5, lane = threadIdx.x & 31;
  int hf = lane >> 4, ln = lane & 15;
  int m0 = wave * 16;
  int rem = win % 648;
  int nhi = (rem % 81) / 9, nwi = rem % 9;   // depth never padded (16 % 2 == 0)

  v8f s[8];
  v16bf a = load_frag(sQ, m0, DH, 0);
  for (int nt = 0; nt < 8; nt++) {
    v16bf bf = load_frag(sK, nt * 16, DH, 0);
    v8f cz = {};
    s[nt] = wmma_bf16(a, bf, cz);
  }
  const float scale = 0.17677669529663687f;  // 32^-0.5
  float cbias[8];
  for (int nt = 0; nt < 8; nt++) {
    int t = nt * 16 + ln;
    bool valid = false;
    if (t < 98) {
      int gh = (t % 49) / 7, gw = t % 7;
      valid = (nhi * 7 + gh < 57) && (nwi * 7 + gw < 57);
    }
    cbias[nt] = valid ? 0.0f : -1000000.0f;
  }
  for (int e = 0; e < 8; e++) {
    float mx = -3.4e38f;
    for (int nt = 0; nt < 8; nt++) {
      s[nt][e] = s[nt][e] * scale + cbias[nt];
      mx = fmaxf(mx, s[nt][e]);
    }
    for (int m = 8; m >= 1; m >>= 1) mx = fmaxf(mx, __shfl_xor(mx, m, 32));
    float sum = 0.0f;
    for (int nt = 0; nt < 8; nt++) { float ev = __expf(s[nt][e] - mx); s[nt][e] = ev; sum += ev; }
    for (int m = 8; m >= 1; m >>= 1) sum += __shfl_xor(sum, m, 32);
    float inv = 1.0f / sum;
    for (int nt = 0; nt < 8; nt++) s[nt][e] *= inv;
  }
  __syncthreads();                    // everyone done with sQ/sK
  for (int nt = 0; nt < 8; nt++)
    for (int e = 0; e < 8; e++)
      sP[(m0 + 8 * hf + e) * NPAD + nt * 16 + ln] = (bf16_t)s[nt][e];
  __syncthreads();

  v8f o[2] = {};
  for (int k0 = 0; k0 < NPAD; k0 += 32) {
    v16bf pa = load_frag(sP, m0, NPAD, k0);
    for (int j = 0; j < 2; j++) {
      v16bf vb = load_frag(sV, j * 16, NPAD, k0);
      o[j] = wmma_bf16(pa, vb, o[j]);
    }
  }
  bf16_t* dst = og + (size_t)win * NPAD * CDIM + head * DH;
  for (int j = 0; j < 2; j++)
    for (int e = 0; e < 8; e++)
      dst[(size_t)(m0 + 8 * hf + e) * CDIM + j * 16 + ln] = (bf16_t)o[j][e];
}

// ---------------------------------------------------------------------------
// Kernel 5: proj GEMM + un-window + crop + residual into d_out (f32).
// ---------------------------------------------------------------------------
__global__ void k_proj(const bf16_t* __restrict__ og, const bf16_t* __restrict__ projT,
                       const float* __restrict__ proj_b, const float* __restrict__ x,
                       float* __restrict__ out) {
  __shared__ bf16_t A[NPAD * CDIM];   // 48 KB
  int win = blockIdx.x;
  const bf16_t* src = og + (size_t)win * NPAD * CDIM;
  for (int i = threadIdx.x; i < NPAD * CDIM / 8; i += 256)
    async_copy_b128(A + i * 8, src + i * 8);
  async_wait0();
  __syncthreads();

  int wave = threadIdx.x >> 5, lane = threadIdx.x & 31;
  int hf = lane >> 4, ln = lane & 15;
  int m0 = wave * 16;
  v8f acc[12] = {};
  for (int k0 = 0; k0 < CDIM; k0 += 32) {
    v16bf a = load_frag(A, m0, CDIM, k0);
    for (int j = 0; j < 12; j++) {
      v16bf bf = load_frag(projT, j * 16, CDIM, k0);
      acc[j] = wmma_bf16(a, bf, acc[j]);
    }
  }
  int bb = win / 648, rem = win % 648, ndi = rem / 81;
  rem %= 81;
  int nhi = rem / 9, nwi = rem % 9;
  for (int e = 0; e < 8; e++) {
    int t = m0 + 8 * hf + e;
    if (t >= 98) continue;
    int gd = t / 49, rr = t % 49, gh = rr / 7, gw = rr % 7;
    int d = ndi * 2 + gd, h = nhi * 7 + gh, w = nwi * 7 + gw;
    if (h >= 57 || w >= 57) continue;
    size_t rowoff = ((((size_t)bb * 16 + d) * 57 + h) * 57 + w) * CDIM;
    for (int j = 0; j < 12; j++) {
      int c = j * 16 + ln;
      out[rowoff + c] = x[rowoff + c] + acc[j][e] + proj_b[c];
    }
  }
}

// ---------------------------------------------------------------------------
// Kernel 6: LN2 + MLP + residual, in-place on d_out (each block owns its rows).
// 128 threads (4 waves), 64 token rows/block; fc2 accumulated in registers.
// ---------------------------------------------------------------------------
__global__ void k_mlp(const float* __restrict__ g2, const float* __restrict__ b2,
                      const bf16_t* __restrict__ fc1T, const float* __restrict__ fc1_b,
                      const bf16_t* __restrict__ fc2T, const float* __restrict__ fc2_b,
                      float* __restrict__ out) {
  __shared__ bf16_t Xl[64 * CDIM];    // 24 KB
  __shared__ bf16_t Hl[64 * 64];      //  8 KB (wave-local 16x64 strips)
  int wave = threadIdx.x >> 5, lane = threadIdx.x & 31;
  int hf = lane >> 4, ln = lane & 15;
  int row0 = blockIdx.x * 64;

  for (int r = 0; r < 16; r++) {
    int lr = wave * 16 + r;
    int grow = row0 + lr;
    if (grow >= TOK) {
      for (int c = lane; c < CDIM; c += 32) Xl[lr * CDIM + c] = (bf16_t)0.0f;
      continue;
    }
    const float* xr = out + (size_t)grow * CDIM;
    float v[6], s = 0.0f;
    for (int j = 0; j < 6; j++) { v[j] = xr[lane + 32 * j]; s += v[j]; }
    for (int m = 16; m >= 1; m >>= 1) s += __shfl_xor(s, m, 32);
    float mean = s * (1.0f / 192.0f), q = 0.0f;
    for (int j = 0; j < 6; j++) { float d2 = v[j] - mean; q += d2 * d2; }
    for (int m = 16; m >= 1; m >>= 1) q += __shfl_xor(q, m, 32);
    float rstd = rsqrtf(q * (1.0f / 192.0f) + 1e-5f);
    for (int j = 0; j < 6; j++) {
      int c = lane + 32 * j;
      Xl[lr * CDIM + c] = (bf16_t)((v[j] - mean) * rstd * g2[c] + b2[c]);
    }
  }
  __syncthreads();

  int m0 = wave * 16;
  v8f acc[12] = {};
  for (int hc = 0; hc < 12; hc++) {   // 12 hidden chunks of 64
    v8f hreg[4] = {};
    for (int k0 = 0; k0 < CDIM; k0 += 32) {
      v16bf a = load_frag(Xl, m0, CDIM, k0);
      for (int j = 0; j < 4; j++) {
        v16bf bf = load_frag(fc1T, hc * 64 + j * 16, CDIM, k0);
        hreg[j] = wmma_bf16(a, bf, hreg[j]);
      }
    }
    for (int j = 0; j < 4; j++)
      for (int e = 0; e < 8; e++) {
        float xv = hreg[j][e] + fc1_b[hc * 64 + j * 16 + ln];
        float ge = 0.5f * xv * (1.0f + erff(xv * 0.70710678f));
        Hl[(m0 + 8 * hf + e) * 64 + j * 16 + ln] = (bf16_t)ge;
      }
    __syncthreads();
    for (int k0 = 0; k0 < 64; k0 += 32) {
      v16bf a = load_frag(Hl, m0, 64, k0);
      for (int j = 0; j < 12; j++) {
        v16bf bf = load_frag(fc2T, j * 16, HID, hc * 64 + k0);
        acc[j] = wmma_bf16(a, bf, acc[j]);
      }
    }
    __syncthreads();
  }
  for (int e = 0; e < 8; e++) {
    int grow = row0 + m0 + 8 * hf + e;
    if (grow >= TOK) continue;
    float* orow = out + (size_t)grow * CDIM;
    for (int j = 0; j < 12; j++) {
      int c = j * 16 + ln;
      orow[c] = orow[c] + acc[j][e] + fc2_b[c];
    }
  }
}

// ---------------------------------------------------------------------------
extern "C" void kernel_launch(void* const* d_in, const int* in_sizes, int n_in,
                              void* d_out, int out_size, void* d_ws, size_t ws_size,
                              hipStream_t stream) {
  (void)in_sizes; (void)n_in; (void)out_size; (void)ws_size;
  const float* x       = (const float*)d_in[0];
  const float* norm1_g = (const float*)d_in[1];
  const float* norm1_b = (const float*)d_in[2];
  const float* qkv_w   = (const float*)d_in[3];
  const float* qkv_b   = (const float*)d_in[4];
  const float* proj_w  = (const float*)d_in[5];
  const float* proj_b  = (const float*)d_in[6];
  const float* norm2_g = (const float*)d_in[7];
  const float* norm2_b = (const float*)d_in[8];
  const float* fc1_w   = (const float*)d_in[9];
  const float* fc1_b   = (const float*)d_in[10];
  const float* fc2_w   = (const float*)d_in[11];
  const float* fc2_b   = (const float*)d_in[12];
  float* out = (float*)d_out;

  char* ws = (char*)d_ws;
  const size_t act = (size_t)NWIN * NPAD * CDIM * sizeof(bf16_t);  // 63.7 MB each
  bf16_t* qkvT = (bf16_t*)(ws);
  bf16_t* projT= (bf16_t*)(ws + 221184);
  bf16_t* fc1T = (bf16_t*)(ws + 294912);
  bf16_t* fc2T = (bf16_t*)(ws + 589824);
  bf16_t* xw   = (bf16_t*)(ws + 884736);
  bf16_t* qb   = (bf16_t*)(ws + 884736 + act);
  bf16_t* kb   = (bf16_t*)(ws + 884736 + 2 * act);
  bf16_t* vTb  = (bf16_t*)(ws + 884736 + 3 * act);
  bf16_t* og   = (bf16_t*)(ws + 884736 + 4 * act);

  k_prep_weights<<<576, 256, 0, stream>>>(qkv_w, proj_w, fc1_w, fc2_w,
                                          qkvT, projT, fc1T, fc2T);
  k_ln1<<<NWIN, 256, 0, stream>>>(x, norm1_g, norm1_b, xw);
  k_qkv<<<NWIN * 9, 256, 0, stream>>>(xw, qkvT, qkv_b, qb, kb, vTb);
  k_attn<<<NWIN * HEADS, 256, 0, stream>>>(qb, kb, vTb, og);
  k_proj<<<NWIN, 256, 0, stream>>>(og, projT, proj_b, x, out);
  k_mlp<<<(TOK + 63) / 64, 128, 0, stream>>>(norm2_g, norm2_b, fc1T, fc1_b,
                                             fc2T, fc2_b, out);
}
